// Tokenizer_22393959481379
// MI455X (gfx1250) — matrix operand order repaired and checked by reading, not compiled, not run on hardware
//
#include <hip/hip_runtime.h>
#include <hip/hip_bf16.h>

typedef __attribute__((ext_vector_type(16))) __bf16 v16bf;
typedef __attribute__((ext_vector_type(8)))  __bf16 v8bf;
typedef __attribute__((ext_vector_type(8)))  float  v8f;

namespace {
constexpr int kB = 16, kT = 2048, kC = 1024, kV = 4096;
constexpr int kM = kB * kT;   // 32768 z rows
constexpr int kN = kV;        // 4095 real codebook entries (e = weight[1:]) + 1 pad column

// ---- output layout (tuple concatenated flat, fp32) ----
constexpr long long NSP_OFF   = 0;          // n_student_probs (B,T,C)
constexpr long long SM_OFF    = 33554432;   // smoothness_loss
constexpr long long CM_OFF    = 33554433;   // commitment_loss
constexpr long long ZQ_OFF    = 33554434;   // z_q (B,T,C)
constexpr long long NZQ_OFF   = 67108866;   // n_z_q (B,T,C)
constexpr long long NMASK_OFF = 100663298;  // n_mask (B,T,1)
constexpr long long SEL_OFF   = 100696066;  // sel (B,T)
constexpr long long SELR_OFF  = 100728834;  // sel_rep (B,T)

// ---- workspace layout (bytes) ----
constexpr size_t WS_Z16  = 0;                                  // bf16 zn: kM*kC*2
constexpr size_t WS_E16  = WS_Z16 + (size_t)kM * kC * 2;       // bf16 e : kN*kC*2
constexpr size_t WS_EN2  = WS_E16 + (size_t)kN * kC * 2;       // ||e||^2: kN*4
constexpr size_t WS_INV  = WS_EN2 + (size_t)kN * 4;            // inv norms: kM*4
constexpr size_t WS_BEST = WS_INV + (size_t)kM * 4;            // packed argmin: kM*8
constexpr size_t WS_ACC  = WS_BEST + (size_t)kM * 8;           // [mask_sum, commit_sum, smooth_sum]
} // namespace

__device__ __forceinline__ float block_reduce_sum(float v, float* sm) {
  int tid = threadIdx.x;
  sm[tid] = v;
  __syncthreads();
  for (int s = blockDim.x >> 1; s > 0; s >>= 1) {
    if (tid < s) sm[tid] += sm[tid + s];
    __syncthreads();
  }
  float r = sm[0];
  __syncthreads();
  return r;
}

// ---- K1: row norms, bf16 zn, best-init, accumulator zeroing ----
__global__ void k_prep_z(const float* __restrict__ z, __bf16* __restrict__ z16,
                         float* __restrict__ invn, unsigned long long* __restrict__ best,
                         float* __restrict__ acc) {
  __shared__ float sm[256];
  const int row = blockIdx.x, tid = threadIdx.x;
  const float* zr = z + (size_t)row * kC;
  float ss = 0.f;
  for (int c = tid; c < kC; c += 256) { float v = zr[c]; ss += v * v; }
  ss = block_reduce_sum(ss, sm);
  const float inv = 1.0f / fmaxf(sqrtf(ss), 1e-12f);
  for (int c = tid; c < kC; c += 256) z16[(size_t)row * kC + c] = (__bf16)(zr[c] * inv);
  if (tid == 0) { invn[row] = inv; best[row] = ~0ULL; }
  if (row == 0 && tid < 3) acc[tid] = 0.f;
}

// ---- K2: bf16 codebook (rows 1..4095 of weight) + ||e||^2, pad col kN-1 ----
__global__ void k_prep_e(const float* __restrict__ w, __bf16* __restrict__ e16,
                         float* __restrict__ en2) {
  __shared__ float sm[256];
  const int j = blockIdx.x, tid = threadIdx.x;
  if (j < kV - 1) {
    const float* er = w + (size_t)(j + 1) * kC;
    float ss = 0.f;
    for (int c = tid; c < kC; c += 256) { float v = er[c]; ss += v * v; }
    ss = block_reduce_sum(ss, sm);
    for (int c = tid; c < kC; c += 256) e16[(size_t)j * kC + c] = (__bf16)er[c];
    if (tid == 0) en2[j] = ss;
  } else {
    for (int c = tid; c < kC; c += 256) e16[(size_t)j * kC + c] = (__bf16)0.0f;
    if (tid == 0) en2[j] = 3.0e38f;  // never wins argmin
  }
}

// fragment loaders (contiguous k order: element s of lane half h is k = 16h + s)
__device__ __forceinline__ v16bf load_frag_a(const __bf16* __restrict__ ap, int k0) {
  union { v16bf v; v8bf h[2]; } au;
  au.h[0] = *(const v8bf*)(ap + k0);
  au.h[1] = *(const v8bf*)(ap + k0 + 8);
  return au.v;
}
__device__ __forceinline__ v16bf load_frag_b(const __bf16* __restrict__ bp, int j, int k0) {
  return *(const v16bf*)(bp + (size_t)j * 16 * kC + k0);
}

// ---- K3: bf16 WMMA GEMM (zn @ e^T) + per-row argmin of ||e||^2 - 2*dot ----
// workgroup = 4 waves = 64(M) x 64(N) tile; wave = 16x64.
// K=1024 in macro-steps of 64: explicit two-stage ping-pong with named register
// buffers so each WMMA stage only waits on its own loads while the other
// stage's loads remain in flight (no dynamic buffer indexing, <256 VGPRs).
__global__ void __launch_bounds__(128) k_gemm_argmin(
    const __bf16* __restrict__ Z, const __bf16* __restrict__ E,
    const float* __restrict__ en2, unsigned long long* __restrict__ best) {
  __shared__ float sd[64][65];
  const int tid  = threadIdx.x;
  const int wave = tid >> 5, lane = tid & 31;
  const int rowBase = blockIdx.y * 64 + wave * 16;
  const int colBase = blockIdx.x * 64;
  const int mr = lane & 15;
  const int hi = lane >> 4;  // 0 or 1 (lane half)

  // A frag (16x32 bf16): lanes 0-15 hold K=0..15, lanes 16-31 K=16..31
  const __bf16* ap = Z + (size_t)(rowBase + mr) * kC + hi * 16;
  // B frag (32x16 bf16, col-major == row of row-major E): lanes 0-15 K=0..15, 16-31 K=16..31
  const __bf16* bp = E + (size_t)(colBase + mr) * kC + hi * 16;

  const v8f vzero = {0.f, 0.f, 0.f, 0.f, 0.f, 0.f, 0.f, 0.f};
  v8f acc[4];
#pragma unroll
  for (int j = 0; j < 4; ++j) acc[j] = vzero;

  // prologue: load chunk 0 into buffer set 0
  v16bf a0 = load_frag_a(ap, 0);
  v16bf b0_0 = load_frag_b(bp, 0, 0);
  v16bf b0_1 = load_frag_b(bp, 1, 0);
  v16bf b0_2 = load_frag_b(bp, 2, 0);
  v16bf b0_3 = load_frag_b(bp, 3, 0);

#pragma unroll 1
  for (int k0 = 0; k0 < kC; k0 += 64) {
    // stage 0: prefetch k0+32 into set 1, then consume set 0
    const int k1 = k0 + 32;
    v16bf a1   = load_frag_a(ap, k1);
    v16bf b1_0 = load_frag_b(bp, 0, k1);
    v16bf b1_1 = load_frag_b(bp, 1, k1);
    v16bf b1_2 = load_frag_b(bp, 2, k1);
    v16bf b1_3 = load_frag_b(bp, 3, k1);
    acc[0] = __builtin_amdgcn_wmma_f32_16x16x32_bf16(false, a0, false, b0_0, (short)0, acc[0], false, false);
    acc[1] = __builtin_amdgcn_wmma_f32_16x16x32_bf16(false, a0, false, b0_1, (short)0, acc[1], false, false);
    acc[2] = __builtin_amdgcn_wmma_f32_16x16x32_bf16(false, a0, false, b0_2, (short)0, acc[2], false, false);
    acc[3] = __builtin_amdgcn_wmma_f32_16x16x32_bf16(false, a0, false, b0_3, (short)0, acc[3], false, false);

    // stage 1: prefetch k0+64 into set 0 (if any), then consume set 1
    const int k2 = k0 + 64;
    if (k2 < kC) {
      a0   = load_frag_a(ap, k2);
      b0_0 = load_frag_b(bp, 0, k2);
      b0_1 = load_frag_b(bp, 1, k2);
      b0_2 = load_frag_b(bp, 2, k2);
      b0_3 = load_frag_b(bp, 3, k2);
    }
    acc[0] = __builtin_amdgcn_wmma_f32_16x16x32_bf16(false, a1, false, b1_0, (short)0, acc[0], false, false);
    acc[1] = __builtin_amdgcn_wmma_f32_16x16x32_bf16(false, a1, false, b1_1, (short)0, acc[1], false, false);
    acc[2] = __builtin_amdgcn_wmma_f32_16x16x32_bf16(false, a1, false, b1_2, (short)0, acc[2], false, false);
    acc[3] = __builtin_amdgcn_wmma_f32_16x16x32_bf16(false, a1, false, b1_3, (short)0, acc[3], false, false);
  }

  // C/D layout: lanes 0-15 -> (M=i, N=lane); lanes 16-31 -> (M=8+i, N=lane-16)
#pragma unroll
  for (int j = 0; j < 4; ++j) {
#pragma unroll
    for (int i = 0; i < 8; ++i) {
      const int mm = i + hi * 8;
      const int nn = (lane & 15) + j * 16;
      sd[wave * 16 + mm][nn] = en2[colBase + nn] - 2.0f * acc[j][i];
    }
  }
  __syncthreads();

  if (tid < 64) {
    float bd = 3.4e38f; int bn = 0;
#pragma unroll 4
    for (int n = 0; n < 64; ++n) {
      const float d = sd[tid][n];
      if (d < bd) { bd = d; bn = n; }   // strict < keeps first index on ties
    }
    const unsigned ub  = __float_as_uint(bd);
    const unsigned key = (ub & 0x80000000u) ? ~ub : (ub | 0x80000000u);
    const unsigned long long packed =
        ((unsigned long long)key << 32) | (unsigned)(colBase + bn);
    atomicMin(best + ((size_t)blockIdx.y * 64 + tid), packed);
  }
}

// ---- K4: z_q gather + commitment-loss partial + mask count ----
__global__ void k_zq_commit(const float* __restrict__ z, const float* __restrict__ w,
                            const float* __restrict__ maskp, const float* __restrict__ invn,
                            const unsigned long long* __restrict__ best,
                            float* __restrict__ out, float* __restrict__ acc) {
  __shared__ float sm[256];
  const int row = blockIdx.x, tid = threadIdx.x;
  const unsigned idx = (unsigned)(best[row] & 0xffffffffu);  // into e = w[1:]
  const float mval = maskp[row];
  const float inv  = invn[row];
  const float* qr  = w + (size_t)(idx + 1) * kC;
  const float* zr  = z + (size_t)row * kC;
  float part = 0.f;
  for (int c = tid; c < kC; c += 256) {
    const float q = qr[c] * mval;                 // z_q forward value
    out[ZQ_OFF + (size_t)row * kC + c] = q;
    const float zn  = zr[c] * inv;
    const float dlt = zn - q;
    part += dlt * dlt * mval;
  }
  part = block_reduce_sum(part, sm);
  if (tid == 0) { atomicAdd(&acc[1], part); atomicAdd(&acc[0], mval); }
}

// ---- K5: smoothness-loss partial ----
__global__ void k_smooth(const float* __restrict__ z, const float* __restrict__ maskp,
                         const float* __restrict__ invn, float* __restrict__ acc) {
  __shared__ float sm[256];
  const int t = blockIdx.x + 1, b = blockIdx.y, tid = threadIdx.x;
  const size_t r0 = (size_t)b * kT + t - 1, r1 = r0 + 1;
  const float mval = maskp[r1];
  const float i0 = invn[r0], i1 = invn[r1];
  const float* z0 = z + r0 * kC;
  const float* z1 = z + r1 * kC;
  float part = 0.f;
  for (int c = tid; c < kC; c += 256) {
    const float d = z0[c] * i0 - z1[c] * i1;
    part += d * d;
  }
  part = block_reduce_sum(part, sm);
  if (tid == 0) atomicAdd(&acc[2], part * mval);
}

// ---- K6: finalize scalar losses ----
__global__ void k_finalize(const float* __restrict__ acc, float* __restrict__ out) {
  const float valid = acc[0] * (float)kC;
  out[SM_OFF] = acc[2] / valid;
  out[CM_OFF] = acc[1] / valid;
}

// ---- K7: per-batch run-length dedup / compaction ----
__global__ void k_dedup(const float* __restrict__ maskp,
                        const unsigned long long* __restrict__ best,
                        float* __restrict__ out) {
  __shared__ int starts[kT];
  __shared__ int s_n;
  const int b = blockIdx.x, tid = threadIdx.x;
  if (tid == 0) {
    int cnt = 0, prev = -1;
    for (int t = 0; t < kT; ++t) {
      const bool mb = maskp[(size_t)b * kT + t] > 0.5f;
      const int  id = (int)(best[(size_t)b * kT + t] & 0xffffffffu);
      if (mb && (t == 0 || id != prev)) starts[cnt++] = t;
      prev = id;
    }
    s_n = cnt;
  }
  __syncthreads();
  const int nr = s_n;

  for (int t = tid; t < kT; t += 256) {
    const bool mb = maskp[(size_t)b * kT + t] > 0.5f;
    const int  id = (int)(best[(size_t)b * kT + t] & 0xffffffffu);
    out[NMASK_OFF + (size_t)b * kT + t] = (t < nr) ? 1.0f : 0.0f;
    float selv = 0.f;
    if (t < nr) {
      const int st = starts[t];
      selv = (float)((int)(best[(size_t)b * kT + st] & 0xffffffffu) + 1);
    }
    out[SEL_OFF  + (size_t)b * kT + t] = selv;
    out[SELR_OFF + (size_t)b * kT + t] = mb ? (float)(id + 1) : 0.f;
  }

  for (int r = 0; r < kT; ++r) {
    const size_t dst = ((size_t)b * kT + r) * kC;
    if (r < nr) {
      const size_t src = ((size_t)b * kT + starts[r]) * kC;
      for (int c = tid; c < kC; c += 256) {
        const float v = out[ZQ_OFF + src + c];
        out[NZQ_OFF + dst + c] = v;
        out[NSP_OFF + dst + c] = v;   // n_student_probs is the same tensor
      }
    } else {
      for (int c = tid; c < kC; c += 256) {
        out[NZQ_OFF + dst + c] = 0.f;
        out[NSP_OFF + dst + c] = 0.f;
      }
    }
  }
}

extern "C" void kernel_launch(void* const* d_in, const int* in_sizes, int n_in,
                              void* d_out, int out_size, void* d_ws, size_t ws_size,
                              hipStream_t stream) {
  const float* z     = (const float*)d_in[0];  // (B,T,C)
  const float* w     = (const float*)d_in[1];  // (V,C)
  const float* maskp = (const float*)d_in[2];  // (B,T,1)
  float* out = (float*)d_out;

  char* ws = (char*)d_ws;
  __bf16* z16 = (__bf16*)(ws + WS_Z16);
  __bf16* e16 = (__bf16*)(ws + WS_E16);
  float*  en2 = (float*)(ws + WS_EN2);
  float*  invn = (float*)(ws + WS_INV);
  unsigned long long* best = (unsigned long long*)(ws + WS_BEST);
  float*  acc = (float*)(ws + WS_ACC);

  k_prep_z<<<kM, 256, 0, stream>>>(z, z16, invn, best, acc);
  k_prep_e<<<kN, 256, 0, stream>>>(w, e16, en2);

  dim3 g3(kN / 64, kM / 64);
  k_gemm_argmin<<<g3, 128, 0, stream>>>(z16, e16, en2, best);

  k_zq_commit<<<kM, 256, 0, stream>>>(z, w, maskp, invn, best, out, acc);

  dim3 g5(kT - 1, kB);
  k_smooth<<<g5, 256, 0, stream>>>(z, maskp, invn, acc);
  k_finalize<<<1, 1, 0, stream>>>(acc, out);

  k_dedup<<<kB, 256, 0, stream>>>(maskp, best, out);
}
